// EGNN_67370857005194
// MI455X (gfx1250) — compile-verified, matrix operand order checked by reading, not run on hardware
//
#include <hip/hip_runtime.h>

typedef __bf16 bf16;
typedef __attribute__((ext_vector_type(16))) __bf16 v16bf;
typedef __attribute__((ext_vector_type(8)))  float  v8f;
typedef __attribute__((ext_vector_type(4)))  unsigned int v4u;

#define N_NODES 16384
#define N_EDGES 262144
#define IN_DIM  512
#define HID     256
#define KE      1088   // 1076 padded to multiple of 32
#define KN      1024
#define TM      64     // edges per workgroup (four 16-row M-tiles)

union FragU { v16bf v; v4u u[2]; };

// Load a 16-lane-group x 32-K bf16 WMMA fragment from [row][K-contiguous] storage.
// Works for A tiles in LDS and for B (= W stored [out][in], out==N, in==K) in global.
__device__ __forceinline__ v16bf frag_row(const bf16* base, int ld, int lane, int k0) {
  int r  = lane & 15;
  int hi = lane >> 4;
  const bf16* p = base + (size_t)r * ld + k0 + hi * 8;
  FragU f;
  f.u[0] = *(const v4u*)(p);
  f.u[1] = *(const v4u*)(p + 16);
  return f.v;
}

__device__ __forceinline__ float silu_f(float x) {
  return x / (1.0f + __expf(-x));
}

// ---------------- prep kernels ----------------

__global__ __launch_bounds__(256) void cvt_pad_kernel(const float* __restrict__ src,
                                                      bf16* __restrict__ dst,
                                                      int O, int I, int Ipad) {
  int idx = blockIdx.x * 256 + threadIdx.x;
  if (idx >= O * Ipad) return;
  int o = idx / Ipad, k = idx - o * Ipad;
  float v = (k < I) ? src[(size_t)o * I + k] : 0.0f;
  dst[idx] = (bf16)v;
}

__global__ __launch_bounds__(256) void zero_f32_kernel(float* __restrict__ p, int n) {
  int i = blockIdx.x * 256 + threadIdx.x;
  if (i < n) p[i] = 0.0f;
}

__global__ __launch_bounds__(256) void copy_f32_kernel(const float* __restrict__ s,
                                                       float* __restrict__ d, int n) {
  int i = blockIdx.x * 256 + threadIdx.x;
  if (i < n) d[i] = s[i];
}

__global__ __launch_bounds__(256) void finite_fix_kernel(float* __restrict__ p, int n) {
  int i = blockIdx.x * 256 + threadIdx.x;
  if (i < n) {
    float v = p[i];
    if (!__builtin_isfinite(v)) p[i] = 0.0f;
  }
}

// ---------------- fused edge kernel ----------------
// Per WG: 64 edges (4 M-tiles), 256 threads = 8 waves, each wave owns 64 output
// columns. B fragments are loaded once per K-step and reused across all 4 M-tiles
// (2:1 wmma:global-load ratio).
// GEMM1: e_in[64,1088] @ We1b -> silu -> t1[64,512]
// GEMM2: t1 @ We2b -> silu -> e ; atomicAdd into agg[row] ; t2 = bf16(e)
// GEMM3: t2 @ Wc1b -> silu -> t3 ; cu = t3 . Wc2 ; clamp ; pos atomics
__global__ __launch_bounds__(256) void egnn_edge_kernel(
    const bf16* __restrict__ hb, const int* __restrict__ ei,
    const float* __restrict__ eattr, const float* __restrict__ pos,
    const bf16* __restrict__ We1, const float* __restrict__ be1,
    const bf16* __restrict__ We2, const float* __restrict__ be2,
    const bf16* __restrict__ Wc1, const float* __restrict__ bc1,
    const float* __restrict__ Wc2,
    float* __restrict__ agg, float* __restrict__ pos_out)
{
  __shared__ __align__(16) bf16 smem[TM * KE];   // 139264 B; later aliased by t1/t2/t3
  __shared__ int   rowS[TM], colS[TM];
  __shared__ float diffS[TM][3], radS[TM], cuacc[TM];

  const int tid  = threadIdx.x;
  const int lane = tid & 31;
  const int wave = tid >> 5;
  const int e0   = blockIdx.x * TM;

  if (tid < TM) {
    int e = e0 + tid;
    int r = ei[e];
    int c = ei[N_EDGES + e];
    rowS[tid] = r; colS[tid] = c;
    float s = 0.f;
#pragma unroll
    for (int d = 0; d < 3; ++d) {
      float df = pos[(size_t)r * 3 + d] - pos[(size_t)c * 3 + d];
      diffS[tid][d] = df;
      s += df * df;
    }
    radS[tid] = fminf(fmaxf(s, 1e-8f), 100.0f);
  }
  __syncthreads();

  // build e_in tile: [h[row](512) | h[col](512) | edge_attr(51) | radial(1) | pad(12)]
  for (int c = tid; c < TM * 128; c += 256) {      // 128 x 16B chunks per row
    int m = c >> 7, j = c & 127;
    const bf16* src = (j < 64) ? (hb + (size_t)rowS[m] * IN_DIM + j * 8)
                               : (hb + (size_t)colS[m] * IN_DIM + (j - 64) * 8);
    *(v4u*)(smem + m * KE + j * 8) = *(const v4u*)src;
  }
  for (int c = tid; c < TM * 64; c += 256) {       // tail 64 bf16 per row
    int m = c >> 6, k2 = c & 63;
    float v = 0.f;
    if (k2 < 51)       v = eattr[(size_t)(e0 + m) * 51 + k2];
    else if (k2 == 51) v = radS[m];
    smem[m * KE + 1024 + k2] = (bf16)v;
  }
  __syncthreads();

  const int nbase = wave * 64;
  const int hi = lane >> 4;
  const int nn = lane & 15;
  bf16* t1 = smem;                 // [64][512] aliases e_in bytes [0, 65536)
  bf16* t2 = smem + TM * IN_DIM;   // [64][512] aliases e_in bytes [65536, 131072)

  // ---- GEMM1 ----
  {
    v8f acc[4][4] = {};
    for (int k0 = 0; k0 < KE; k0 += 32) {
      v16bf b[4];
#pragma unroll
      for (int s = 0; s < 4; ++s)
        b[s] = frag_row(We1 + (size_t)(nbase + s * 16) * KE, KE, lane, k0);
#pragma unroll
      for (int mt = 0; mt < 4; ++mt) {
        v16bf a = frag_row(smem + (size_t)mt * 16 * KE, KE, lane, k0);
#pragma unroll
        for (int s = 0; s < 4; ++s)
          acc[mt][s] = __builtin_amdgcn_wmma_f32_16x16x32_bf16(
              false, a, false, b[s], (short)0, acc[mt][s], false, false);
      }
    }
    __syncthreads();   // e_in dead; safe to overwrite with t1
#pragma unroll
    for (int mt = 0; mt < 4; ++mt)
#pragma unroll
      for (int s = 0; s < 4; ++s) {
        int n = nbase + s * 16 + nn;
        float bias = be1[n];
#pragma unroll
        for (int r = 0; r < 8; ++r) {
          int m = mt * 16 + r + 8 * hi;
          t1[m * IN_DIM + n] = (bf16)silu_f(acc[mt][s][r] + bias);
        }
      }
    __syncthreads();
  }

  // ---- GEMM2 ----
  {
    v8f acc[4][4] = {};
    for (int k0 = 0; k0 < IN_DIM; k0 += 32) {
      v16bf b[4];
#pragma unroll
      for (int s = 0; s < 4; ++s)
        b[s] = frag_row(We2 + (size_t)(nbase + s * 16) * IN_DIM, IN_DIM, lane, k0);
#pragma unroll
      for (int mt = 0; mt < 4; ++mt) {
        v16bf a = frag_row(t1 + (size_t)mt * 16 * IN_DIM, IN_DIM, lane, k0);
#pragma unroll
        for (int s = 0; s < 4; ++s)
          acc[mt][s] = __builtin_amdgcn_wmma_f32_16x16x32_bf16(
              false, a, false, b[s], (short)0, acc[mt][s], false, false);
      }
    }
    __syncthreads();
#pragma unroll
    for (int mt = 0; mt < 4; ++mt)
#pragma unroll
      for (int s = 0; s < 4; ++s) {
        int n = nbase + s * 16 + nn;
        float bias = be2[n];
#pragma unroll
        for (int r = 0; r < 8; ++r) {
          int m = mt * 16 + r + 8 * hi;
          float x = silu_f(acc[mt][s][r] + bias);
          atomicAdd(&agg[(size_t)rowS[m] * IN_DIM + n], x);  // segment_sum(e, row)
          t2[m * IN_DIM + n] = (bf16)x;
        }
      }
    __syncthreads();
  }

  // ---- GEMM3 (coord MLP hidden) ----
  {
    v8f acc[4][4] = {};
    for (int k0 = 0; k0 < IN_DIM; k0 += 32) {
      v16bf b[4];
#pragma unroll
      for (int s = 0; s < 4; ++s)
        b[s] = frag_row(Wc1 + (size_t)(nbase + s * 16) * IN_DIM, IN_DIM, lane, k0);
#pragma unroll
      for (int mt = 0; mt < 4; ++mt) {
        v16bf a = frag_row(t2 + (size_t)mt * 16 * IN_DIM, IN_DIM, lane, k0);
#pragma unroll
        for (int s = 0; s < 4; ++s)
          acc[mt][s] = __builtin_amdgcn_wmma_f32_16x16x32_bf16(
              false, a, false, b[s], (short)0, acc[mt][s], false, false);
      }
    }
    __syncthreads();
    bf16* t3 = smem;   // aliases t1 region (dead after GEMM2)
#pragma unroll
    for (int mt = 0; mt < 4; ++mt)
#pragma unroll
      for (int s = 0; s < 4; ++s) {
        int n = nbase + s * 16 + nn;
        float bias = bc1[n];
#pragma unroll
        for (int r = 0; r < 8; ++r) {
          int m = mt * 16 + r + 8 * hi;
          t3[m * IN_DIM + n] = (bf16)silu_f(acc[mt][s][r] + bias);
        }
      }
  }
  if (tid < TM) cuacc[tid] = 0.0f;
  __syncthreads();

  // cu = t3 . Wc2  (4 threads per edge, 128 K each)
  {
    const bf16* t3 = smem;
    int m = tid >> 2, p = tid & 3;
    float part = 0.f;
#pragma unroll 4
    for (int j = 0; j < 128; ++j) {
      int k = p * 128 + j;
      part += (float)t3[m * IN_DIM + k] * Wc2[k];
    }
    atomicAdd(&cuacc[m], part);
  }
  __syncthreads();

  if (tid < TM) {
    float cu = fminf(fmaxf(cuacc[tid], -1.0f), 1.0f);
    int r = rowS[tid];
#pragma unroll
    for (int d = 0; d < 3; ++d) {
      float tr = diffS[tid][d] * cu;
      if (!__builtin_isfinite(tr)) tr = 0.0f;
      atomicAdd(&pos_out[(size_t)r * 3 + d], tr);
    }
  }
}

// ---------------- node kernel ----------------
// Per WG: 16 nodes; x_in[16,1024] = [hb | bf16(agg)]; two WMMA GEMMs.
__global__ __launch_bounds__(256) void egnn_node_kernel(
    const bf16* __restrict__ hb, const float* __restrict__ agg,
    const bf16* __restrict__ Wn1, const float* __restrict__ bn1,
    const bf16* __restrict__ Wn2, const float* __restrict__ bn2,
    float* __restrict__ hout)
{
  __shared__ __align__(16) bf16 smem[16 * KN];
  const int tid  = threadIdx.x;
  const int lane = tid & 31;
  const int wave = tid >> 5;
  const int m0   = blockIdx.x * 16;

  for (int c = tid; c < 16 * 64; c += 256) {       // h half, 16B chunks
    int m = c >> 6, j = c & 63;
    *(v4u*)(smem + m * KN + j * 8) = *(const v4u*)(hb + (size_t)(m0 + m) * IN_DIM + j * 8);
  }
  for (int c = tid; c < 16 * IN_DIM; c += 256) {   // agg half, f32->bf16
    int m = c >> 9, k = c & 511;
    smem[m * KN + IN_DIM + k] = (bf16)agg[(size_t)(m0 + m) * IN_DIM + k];
  }
  __syncthreads();

  const int nbase = wave * 32;
  const int hi = lane >> 4;
  const int nn = lane & 15;

  v8f acc1[2] = {};
  for (int k0 = 0; k0 < KN; k0 += 32) {
    v16bf a = frag_row(smem, KN, lane, k0);
#pragma unroll
    for (int s = 0; s < 2; ++s) {
      v16bf b = frag_row(Wn1 + (size_t)(nbase + s * 16) * KN, KN, lane, k0);
      acc1[s] = __builtin_amdgcn_wmma_f32_16x16x32_bf16(false, a, false, b,
                                                        (short)0, acc1[s], false, false);
    }
  }
  __syncthreads();   // x_in dead
  bf16* t = smem;    // [16][256]
#pragma unroll
  for (int s = 0; s < 2; ++s) {
    int n = nbase + s * 16 + nn;
    float bias = bn1[n];
#pragma unroll
    for (int r = 0; r < 8; ++r) {
      int m = r + 8 * hi;
      t[m * HID + n] = (bf16)silu_f(acc1[s][r] + bias);
    }
  }
  __syncthreads();

  v8f acc2[2] = {};
  for (int k0 = 0; k0 < HID; k0 += 32) {
    v16bf a = frag_row(t, HID, lane, k0);
#pragma unroll
    for (int s = 0; s < 2; ++s) {
      v16bf b = frag_row(Wn2 + (size_t)(nbase + s * 16) * HID, HID, lane, k0);
      acc2[s] = __builtin_amdgcn_wmma_f32_16x16x32_bf16(false, a, false, b,
                                                        (short)0, acc2[s], false, false);
    }
  }
#pragma unroll
  for (int s = 0; s < 2; ++s) {
    int n = nbase + s * 16 + nn;
    float bias = bn2[n];
#pragma unroll
    for (int r = 0; r < 8; ++r) {
      int m = r + 8 * hi;
      hout[(size_t)(m0 + m) * HID + n] = acc2[s][r] + bias;
    }
  }
}

// ---------------- launch ----------------

extern "C" void kernel_launch(void* const* d_in, const int* in_sizes, int n_in,
                              void* d_out, int out_size, void* d_ws, size_t ws_size,
                              hipStream_t stream) {
  (void)in_sizes; (void)n_in; (void)out_size; (void)ws_size;
  const float* h    = (const float*)d_in[0];
  const int*   ei   = (const int*)d_in[1];
  const float* eatt = (const float*)d_in[2];
  const float* pos  = (const float*)d_in[3];
  const float* We1  = (const float*)d_in[4];
  const float* be1  = (const float*)d_in[5];
  const float* We2  = (const float*)d_in[6];
  const float* be2  = (const float*)d_in[7];
  const float* Wc1  = (const float*)d_in[8];
  const float* bc1  = (const float*)d_in[9];
  const float* Wc2  = (const float*)d_in[10];
  const float* Wn1  = (const float*)d_in[11];
  const float* bn1  = (const float*)d_in[12];
  const float* Wn2  = (const float*)d_in[13];
  const float* bn2  = (const float*)d_in[14];

  char* ws = (char*)d_ws;
  size_t off = 0;
  float* agg  = (float*)(ws + off); off += (size_t)N_NODES * IN_DIM * 4;
  bf16*  hb   = (bf16*) (ws + off); off += (size_t)N_NODES * IN_DIM * 2;
  bf16*  We1b = (bf16*) (ws + off); off += (size_t)IN_DIM * KE * 2;
  bf16*  We2b = (bf16*) (ws + off); off += (size_t)IN_DIM * IN_DIM * 2;
  bf16*  Wc1b = (bf16*) (ws + off); off += (size_t)IN_DIM * IN_DIM * 2;
  bf16*  Wn1b = (bf16*) (ws + off); off += (size_t)HID * KN * 2;
  bf16*  Wn2b = (bf16*) (ws + off); off += (size_t)HID * HID * 2;

  float* hout    = (float*)d_out;
  float* pos_out = hout + (size_t)N_NODES * HID;

  auto nb = [](int n) { return (n + 255) / 256; };

  cvt_pad_kernel<<<nb(IN_DIM * KE),      256, 0, stream>>>(We1, We1b, IN_DIM, 1076, KE);
  cvt_pad_kernel<<<nb(IN_DIM * IN_DIM),  256, 0, stream>>>(We2, We2b, IN_DIM, IN_DIM, IN_DIM);
  cvt_pad_kernel<<<nb(IN_DIM * IN_DIM),  256, 0, stream>>>(Wc1, Wc1b, IN_DIM, IN_DIM, IN_DIM);
  cvt_pad_kernel<<<nb(HID * KN),         256, 0, stream>>>(Wn1, Wn1b, HID, KN, KN);
  cvt_pad_kernel<<<nb(HID * HID),        256, 0, stream>>>(Wn2, Wn2b, HID, HID, HID);
  cvt_pad_kernel<<<nb(N_NODES * IN_DIM), 256, 0, stream>>>(h, hb, N_NODES, IN_DIM, IN_DIM);
  zero_f32_kernel<<<nb(N_NODES * IN_DIM), 256, 0, stream>>>(agg, N_NODES * IN_DIM);
  copy_f32_kernel<<<nb(N_NODES * 3),      256, 0, stream>>>(pos, pos_out, N_NODES * 3);

  egnn_edge_kernel<<<N_EDGES / TM, 256, 0, stream>>>(hb, ei, eatt, pos,
                                                     We1b, be1, We2b, be2,
                                                     Wc1b, bc1, Wc2,
                                                     agg, pos_out);

  finite_fix_kernel<<<nb(N_NODES * 3), 256, 0, stream>>>(pos_out, N_NODES * 3);

  egnn_node_kernel<<<N_NODES / 16, 256, 0, stream>>>(hb, agg, Wn1b, bn1, Wn2b, bn2, hout);
}
